// graph_conv_28991029248103
// MI455X (gfx1250) — matrix operand order repaired and checked by reading, not compiled, not run on hardware
//
#include <hip/hip_runtime.h>

typedef __attribute__((ext_vector_type(16))) __bf16 v16bf;
typedef __attribute__((ext_vector_type(8)))  float  v8f;

#define BB   32
#define TT_T 128
#define VV   128
#define FF   64
#define OO   64
#define KCAT 192      // [X | Y1 | Y2] concatenated K dimension
#define TCHUNK 8      // timesteps per workgroup
#define APITCH 136    // sA row pitch in bf16 (272 B: bank-conflict-free)
#define GPITCH 200    // sG row pitch in bf16 (400 B: bank-conflict-free)

union BFrag { uint4 q[2]; v16bf v; };

__device__ __forceinline__ v16bf load_bfrag(const uint4* h0, const uint4* h1, int idx) {
  BFrag t; t.q[0] = h0[idx]; t.q[1] = h1[idx]; return t.v;
}

// ---------------------------------------------------------------------------
// Prep 1: deg[b][j] = sum_i W[b][i][j]   (column sums, K.sum(W, axis=1))
// ---------------------------------------------------------------------------
__global__ void deg_kernel(const float* __restrict__ W, float* __restrict__ deg) {
  int b = blockIdx.x, j = threadIdx.x;
  const float* Wb = W + (size_t)b * VV * VV;
  float s = 0.f;
  for (int i = 0; i < VV; ++i) s += Wb[i * VV + j];
  deg[b * VV + j] = s;
}

// ---------------------------------------------------------------------------
// Prep 2: A1 = T1^T, A2 = T2^T in bf16 (row = u, col = v).
//   T1[v][u] = (v==u)*(deg[b,v]-1) - W[b,v,u]
//   T2[v][u] = 2*T1[v][u]^2 - (v==u)           (elementwise, per reference)
// ---------------------------------------------------------------------------
__global__ void cheb_kernel(const float* __restrict__ W, const float* __restrict__ deg,
                            __bf16* __restrict__ A1, __bf16* __restrict__ A2) {
  int bu = blockIdx.x;
  int b = bu / VV, u = bu % VV, v = threadIdx.x;
  float w  = W[((size_t)b * VV + v) * VV + u];
  float lt = ((u == v) ? (deg[b * VV + v] - 1.f) : 0.f) - w;
  float t2 = 2.f * lt * lt - ((u == v) ? 1.f : 0.f);
  size_t idx = ((size_t)b * VV + u) * VV + v;
  A1[idx] = (__bf16)lt;
  A2[idx] = (__bf16)t2;
}

// ---------------------------------------------------------------------------
// Prep 3: Thetacat (192x64) f32 -> bf16, stored PRE-SWIZZLED in split-half
// wave32 B-fragment layout: half h (element e = h*8 + e7 of the fragment),
// fragment q = kt*4+nt, lane: Tc[h*6144 + (q*32+lane)*8 + e7] =
//   Theta[kt*32 + (lane>>4)*16 + e][nt*16 + (lane&15)].
// Each half is one lane-contiguous ds_load_b128 (16 B/lane: conflict-free).
// ---------------------------------------------------------------------------
__global__ void theta_kernel(const float* __restrict__ Th, __bf16* __restrict__ Tc) {
  int i = blockIdx.x * blockDim.x + threadIdx.x;   // exactly 12288
  int h    = i / 6144;
  int r    = i % 6144;
  int q    = r >> 8;                               // 0..23 = kk*4 + nt
  int lane = (r >> 3) & 31;
  int e    = h * 8 + (r & 7);
  int kt = q >> 2, nt = q & 3;
  int krow = kt * 32 + (lane >> 4) * 16 + e;
  int col  = nt * 16 + (lane & 15);
  Tc[i] = (__bf16)Th[krow * OO + col];
}

// ---------------------------------------------------------------------------
// Main fused kernel, software-pipelined over timesteps:
//   commit X[t] to LDS -> issue global loads of X[t+1] -> stage1 -> stage2
// A fragments hoisted & reused across 4 N-tiles; all LDS layouts padded /
// split for conflict-free banking.
// ---------------------------------------------------------------------------
__launch_bounds__(256, 2)
__global__ void gconv_kernel(const float* __restrict__ x,
                             const __bf16* __restrict__ A1g,
                             const __bf16* __restrict__ A2g,
                             const __bf16* __restrict__ Tcg,
                             float* __restrict__ out) {
  extern __shared__ char smem_raw[];
  __bf16* sA1  = (__bf16*)smem_raw;                 // 128*136*2 = 34816 B
  __bf16* sA2  = (__bf16*)(smem_raw + 34816);       // 34816 B
  __bf16* sTh  = (__bf16*)(smem_raw + 69632);       // 2 halves x 12288 B
  __bf16* sG   = (__bf16*)(smem_raw + 94208);       // 128*200*2 = 51200 B
  __bf16* sXB0 = (__bf16*)(smem_raw + 145408);      // 8192 B (frag half 0)
  __bf16* sXB1 = (__bf16*)(smem_raw + 153600);      // 8192 B (frag half 1)
  const uint4* sTh0v = (const uint4*)sTh;           // 24 frags x 32 lanes
  const uint4* sTh1v = (const uint4*)(sTh + 6144);
  const uint4* sXB0v = (const uint4*)sXB0;          // 16 frags x 32 lanes
  const uint4* sXB1v = (const uint4*)sXB1;

  const int b    = blockIdx.x >> 4;                 // T/TCHUNK = 16 chunks
  const int t0   = (blockIdx.x & 15) * TCHUNK;
  const int tid  = threadIdx.x;
  const int lane = tid & 31;
  const int wave = tid >> 5;
  const int m    = lane & 15;      // row (A) or column (B/D) within tile
  const int kb   = lane >> 4;      // which K-half this lane holds

  // --- cooperative load: A1/A2 rows into padded LDS, swizzled Theta -------
  {
    const uint4* s1 = (const uint4*)(A1g + (size_t)b * VV * VV);
    const uint4* s2 = (const uint4*)(A2g + (size_t)b * VV * VV);
    for (int i = tid; i < VV * VV / 8; i += 256) {  // 2048: 8 uniform iters
      int row = i >> 4, c = i & 15;
      ((uint4*)(sA1 + row * APITCH))[c] = s1[i];
      ((uint4*)(sA2 + row * APITCH))[c] = s2[i];
    }
    const uint4* st = (const uint4*)Tcg; uint4* dt = (uint4*)sTh;
    for (int i = tid; i < KCAT * OO / 8; i += 256) dt[i] = st[i];
  }

  // --- pipeline prologue: pull X[t0] into registers (32 floats/thread) ----
  float4 xreg[8];
  {
    const float4* src = (const float4*)(x + ((size_t)(b * TT_T + t0)) * VV * FF);
    #pragma unroll
    for (int r = 0; r < 8; ++r) xreg[r] = src[tid + 256 * r];
  }
  __syncthreads();

  for (int tt = 0; tt < TCHUNK; ++tt) {
    const int t = t0 + tt;

    // --- commit X[t]: row-major bf16 into G[:,0:64] AND split-swizzled sXB
    #pragma unroll
    for (int r = 0; r < 8; ++r) {
      const int base = (tid + 256 * r) * 4;
      const int v  = base >> 6;               // 0..127
      const int f0 = base & 63;               // multiple of 4
      const __bf16 e0 = (__bf16)xreg[r].x, e1 = (__bf16)xreg[r].y;
      const __bf16 e2 = (__bf16)xreg[r].z, e3 = (__bf16)xreg[r].w;
      __bf16* g = sG + v * GPITCH + f0;
      g[0] = e0; g[1] = e1; g[2] = e2; g[3] = e3;
      const int kt = v >> 5, kbv = (v >> 4) & 1, e = v & 15;
      __bf16* xb = ((e >> 3) ? sXB1 : sXB0);  // which fragment half
      const int e7 = e & 7;
      #pragma unroll
      for (int q = 0; q < 4; ++q) {
        const int f = f0 + q;
        const int idx = ((kt * 4 + (f >> 4)) * 32 + kbv * 16 + (f & 15)) * 8 + e7;
        xb[idx] = (q == 0) ? e0 : (q == 1) ? e1 : (q == 2) ? e2 : e3;
      }
    }
    __syncthreads();

    // --- issue global loads for X[t+1] (consumed next iteration) ---------
    {
      const int tn = (tt + 1 < TCHUNK) ? (t + 1) : t;   // uniform, in-bounds
      const float4* src = (const float4*)(x + ((size_t)(b * TT_T + tn)) * VV * FF);
      #pragma unroll
      for (int r = 0; r < 8; ++r) xreg[r] = src[tid + 256 * r];
    }

    // --- stage 1: Y1 = A1*X, Y2 = A2*X ; 64 tiles, 2 groups of 4 per wave -
    for (int g = 0; g < 2; ++g) {
      const int idx0 = wave * 8 + g * 4;           // group base, wave-uniform
      const __bf16* Asrc = (idx0 < 32) ? sA1 : sA2;
      const int ut = (idx0 & 31) >> 2;             // shared by the 4 tiles
      v16bf afr[4];                                // hoisted A frags (K=128)
      #pragma unroll
      for (int kk = 0; kk < 4; ++kk) {
        const __bf16* arow = Asrc + (ut * 16 + m) * APITCH + kk * 32;
        #pragma unroll
        for (int j = 0; j < 8; ++j) {              // one contiguous 32B run
          int k = kb * 16 + 2 * j;
          afr[kk][2 * j]     = arow[k];
          afr[kk][2 * j + 1] = arow[k + 1];
        }
      }
      #pragma unroll
      for (int ft = 0; ft < 4; ++ft) {             // 4 N-tiles reuse afr[]
        v8f acc = {};
        #pragma unroll
        for (int kk = 0; kk < 4; ++kk) {
          const v16bf bfrag = load_bfrag(sXB0v, sXB1v, (kk * 4 + ft) * 32 + lane);
          acc = __builtin_amdgcn_wmma_f32_16x16x32_bf16(
                    false, afr[kk], false, bfrag, (short)0, acc, false, false);
        }
        const int colbase = ((idx0 < 32) ? 64 : 128) + ft * 16 + m;
        #pragma unroll
        for (int r = 0; r < 8; ++r)                // D layout: M = r + 8*kb
          sG[(ut * 16 + r + kb * 8) * GPITCH + colbase] = (__bf16)acc[r];
      }
    }
    __syncthreads();

    // --- stage 2: out = relu(G(128x192) * Thetacat(192x64)) ---------------
    {
      const int ut = wave;                         // all 4 tiles share ut
      v16bf afr[6];                                // hoisted A frags (K=192)
      #pragma unroll
      for (int kk = 0; kk < 6; ++kk) {
        const __bf16* grow = sG + (ut * 16 + m) * GPITCH + kk * 32;
        #pragma unroll
        for (int j = 0; j < 8; ++j) {
          int k = kb * 16 + 2 * j;
          afr[kk][2 * j]     = grow[k];
          afr[kk][2 * j + 1] = grow[k + 1];
        }
      }
      #pragma unroll
      for (int ot = 0; ot < 4; ++ot) {             // 4 N-tiles reuse afr[]
        v8f acc = {};
        #pragma unroll
        for (int kk = 0; kk < 6; ++kk) {
          const v16bf bfrag = load_bfrag(sTh0v, sTh1v, (kk * 4 + ot) * 32 + lane);
          acc = __builtin_amdgcn_wmma_f32_16x16x32_bf16(
                    false, afr[kk], false, bfrag, (short)0, acc, false, false);
        }
        float* obase = out + (((size_t)(b * TT_T + t)) * VV + ut * 16) * OO
                           + ot * 16 + m;
        #pragma unroll
        for (int r = 0; r < 8; ++r) {
          float vfin = acc[r];
          obase[(size_t)(r + kb * 8) * OO] = vfin > 0.f ? vfin : 0.f;
        }
      }
    }
    __syncthreads();   // protect G/sXB before next timestep overwrites them
  }
}

// ---------------------------------------------------------------------------
extern "C" void kernel_launch(void* const* d_in, const int* in_sizes, int n_in,
                              void* d_out, int out_size, void* d_ws, size_t ws_size,
                              hipStream_t stream) {
  const float* x     = (const float*)d_in[0];   // (32,128,128,64)
  const float* W     = (const float*)d_in[1];   // (32,128,128)
  const float* Theta = (const float*)d_in[2];   // (3,64,64)
  float* out = (float*)d_out;                   // (32,128,128,64)

  // workspace carve (16B aligned): deg | A1 | A2 | Thetacat  (~2.1 MB)
  char* ws = (char*)d_ws;
  float*  deg = (float*)ws;                                    // 16 KB
  __bf16* A1  = (__bf16*)(ws + 16384);                         // 1 MB
  __bf16* A2  = A1 + (size_t)BB * VV * VV;                     // 1 MB
  __bf16* Tc  = A2 + (size_t)BB * VV * VV;                     // 24 KB

  deg_kernel  <<<BB,        128, 0, stream>>>(W, deg);
  cheb_kernel <<<BB * VV,   128, 0, stream>>>(W, deg, A1, A2);
  theta_kernel<<<48,        256, 0, stream>>>(Theta, Tc);

  const size_t smem = 161792;                   // 158 KB -> 2 blocks / WGP
  gconv_kernel<<<BB * (TT_T / TCHUNK), 256, smem, stream>>>(x, A1, A2, Tc, out);
}
